// SparseMLPWithLoRA_38603166056858
// MI455X (gfx1250) — matrix operand order repaired and checked by reading, not compiled, not run on hardware
//
#include <hip/hip_runtime.h>
#include <hip/hip_bf16.h>

#define HDIM 2048
#define FFP  1024
#define NE   8
#define RLORA 16
#define KC   32
#define MT   16
#define LORA_SCALE 2.0f

typedef __attribute__((ext_vector_type(16))) __bf16 v16bf;
typedef __attribute__((ext_vector_type(8)))  float  v8f;

// float -> bf16 bits, round-to-nearest-even (used by the prep kernels and for mids/ts in moe_mlp_kernel)
static __device__ __forceinline__ unsigned short f2bf(float f) {
  unsigned int u = __float_as_uint(f);
  u += 0x7fffu + ((u >> 16) & 1u);
  return (unsigned short)(u >> 16);
}

// A-operand fragment (bf16 16x32) from a row-major K-contiguous row:
// lane = {m = lane&15, hi = lane>>4}; halves 0..7 = K[kbase+8*hi ..],
// halves 8..15 = K[kbase+16+8*hi ..]   (two 16B loads)
static __device__ __forceinline__ v16bf a_frag(const unsigned short* row, int kbase, int hi) {
  v16bf f;
  ((uint4*)&f)[0] = *(const uint4*)(row + kbase + 8 * hi);
  ((uint4*)&f)[1] = *(const uint4*)(row + kbase + 16 + 8 * hi);
  return f;
}

// B-operand fragment (bf16 32x16) from pre-transposed [n][k] storage:
// lane = {n = lane&15, hi = lane>>4}; halves 0..7 = K[kbase+8*hi ..],
// halves 8..15 = K[kbase+16+8*hi ..]  (two 16B loads)
static __device__ __forceinline__ v16bf b_frag(const unsigned short* row, int kbase, int hi) {
  v16bf f;
  ((uint4*)&f)[0] = *(const uint4*)(row + kbase + 8 * hi);
  ((uint4*)&f)[1] = *(const uint4*)(row + kbase + 16 + 8 * hi);
  return f;
}

static __device__ __forceinline__ v8f wmma_bf16(v16bf a, v16bf b, v8f c) {
  return __builtin_amdgcn_wmma_f32_16x16x32_bf16(false, a, false, b, (short)0, c, false, false);
}

// ---------------- one-time prep: elementwise f32 -> bf16 ----------------
__global__ __launch_bounds__(256)
void cvt_bf16_kernel(const float* __restrict__ in, unsigned short* __restrict__ out,
                     long long n4) {
  long long i = (long long)blockIdx.x * 256 + threadIdx.x;  // index over groups of 4
  if (i >= n4) return;
  float4 v = ((const float4*)in)[i];
  uint2 o;
  o.x = (unsigned)f2bf(v.x) | ((unsigned)f2bf(v.y) << 16);
  o.y = (unsigned)f2bf(v.z) | ((unsigned)f2bf(v.w) << 16);
  ((uint2*)out)[i] = o;
}

// ---------------- one-time prep: batched transpose + convert ----------------
// in: [batch][rows][cols] f32  ->  out: [batch][cols][rows] bf16
__global__ __launch_bounds__(256)
void transpose_cvt_kernel(const float* __restrict__ in, unsigned short* __restrict__ out,
                          int rows, int cols) {
  __shared__ float tile[32][33];
  const int b = blockIdx.z;
  const float* inb = in + (size_t)b * rows * cols;
  unsigned short* outb = out + (size_t)b * rows * cols;
  const int tx = threadIdx.x;  // 0..31
  const int ty = threadIdx.y;  // 0..7
  const int c0 = blockIdx.x * 32;
  const int r0 = blockIdx.y * 32;
  #pragma unroll
  for (int j = 0; j < 32; j += 8) {
    int r = r0 + ty + j, c = c0 + tx;
    tile[ty + j][tx] = (r < rows && c < cols) ? inb[(size_t)r * cols + c] : 0.f;
  }
  __syncthreads();
  #pragma unroll
  for (int j = 0; j < 32; j += 8) {
    int c = c0 + ty + j, r = r0 + tx;
    if (c < cols && r < rows) outb[(size_t)c * rows + r] = f2bf(tile[tx][ty + j]);
  }
}

// ---------------- one-time prep: B^T with K padded to 32 (zeros) ----------------
// B: [E][R][H] f32 -> Bt: [E][H][32] bf16, Bt[e][h][r] = B[e][r][h], r>=16 -> 0
__global__ __launch_bounds__(256)
void btpad_kernel(const float* __restrict__ B, unsigned short* __restrict__ Bt) {
  int idx = blockIdx.x * 256 + threadIdx.x;  // e*HDIM + h
  if (idx >= NE * HDIM) return;
  int e = idx / HDIM, h = idx % HDIM;
  const float* src = B + (size_t)e * RLORA * HDIM + h;
  unsigned short* dst = Bt + (size_t)idx * 32;
  #pragma unroll
  for (int r = 0; r < RLORA; ++r) dst[r] = f2bf(src[(size_t)r * HDIM]);
  #pragma unroll
  for (int r = RLORA; r < 32; ++r) dst[r] = 0;
}

// ---------------- gating: softmax + top-2 + compaction ----------------
__global__ __launch_bounds__(256)
void gate_topk_kernel(const float* __restrict__ x, const float* __restrict__ gw,
                      int* __restrict__ cnt, int* __restrict__ tokL,
                      float* __restrict__ wgtL, int T) {
  int t = blockIdx.x * blockDim.x + threadIdx.x;
  if (t >= T) return;
  float lg[NE];
  #pragma unroll
  for (int e = 0; e < NE; ++e) lg[e] = 0.f;
  const float* xr = x + (size_t)t * HDIM;
  for (int h = 0; h < HDIM; h += 4) {
    float4 xv = *(const float4*)(xr + h);
    #pragma unroll
    for (int e = 0; e < NE; ++e) {
      lg[e] += xv.x * gw[(h + 0) * NE + e] + xv.y * gw[(h + 1) * NE + e]
             + xv.z * gw[(h + 2) * NE + e] + xv.w * gw[(h + 3) * NE + e];
    }
  }
  float mx = lg[0];
  #pragma unroll
  for (int e = 1; e < NE; ++e) mx = fmaxf(mx, lg[e]);
  float p[NE];
  #pragma unroll
  for (int e = 0; e < NE; ++e) p[e] = __expf(lg[e] - mx);
  int i1 = 0; float v1 = p[0];
  #pragma unroll
  for (int e = 1; e < NE; ++e) if (p[e] > v1) { v1 = p[e]; i1 = e; }
  int i2 = (i1 == 0) ? 1 : 0; float v2 = p[i2];
  #pragma unroll
  for (int e = 0; e < NE; ++e) if (e != i1 && p[e] > v2) { v2 = p[e]; i2 = e; }
  float denom = fmaxf(v1 + v2, 1e-30f);
  float w1 = v1 / denom, w2 = v2 / denom;
  int q1 = atomicAdd(&cnt[i1], 1);
  tokL[(size_t)i1 * T + q1] = t; wgtL[(size_t)i1 * T + q1] = w1;
  int q2 = atomicAdd(&cnt[i2], 1);
  tokL[(size_t)i2 * T + q2] = t; wgtL[(size_t)i2 * T + q2] = w2;
}

// ---------------- expert MLP + LoRA: bf16 WMMA grouped GEMM ----------------
// Pre-converted/pre-transposed operands; inner loops are software-pipelined
// global_load_b128 -> v_wmma_f32_16x16x32_bf16 (next tile's loads issued
// before the current tile's WMMA so the wait is partial, not a full drain).
__global__ __launch_bounds__(256)
void moe_mlp_kernel(const unsigned short* __restrict__ xbf,  // [T][H]
                    const unsigned short* __restrict__ Wgt,  // [E][FFP][H]
                    const unsigned short* __restrict__ Wut,  // [E][FFP][H]
                    const unsigned short* __restrict__ Wdt,  // [E][H][FFP]
                    const unsigned short* __restrict__ At,   // [E][R][H]
                    const unsigned short* __restrict__ Bt,   // [E][H][32]
                    const int* __restrict__ cnt, const int* __restrict__ tokL,
                    const float* __restrict__ wgtL,
                    float* __restrict__ out, int T) {
  __shared__ __align__(16) unsigned short mids[MT][FFP];  // 32 KB, silu(g)*u bf16
  __shared__ __align__(16) unsigned short ts[MT][KC];     //  1 KB, LoRA t, K-padded
  __shared__ int   stok[MT];
  __shared__ float swv[MT];

  const int e     = blockIdx.y;
  const int nTok  = cnt[e];
  const int tile0 = blockIdx.x * MT;
  if (tile0 >= nTok) return;

  const int tid  = threadIdx.x;
  const int wave = tid >> 5;
  const int lane = tid & 31;
  const int l16  = lane & 15;
  const int hi   = lane >> 4;

  if (tid < MT) {
    int idx = tile0 + tid;
    int tk = 0; float wv = 0.f;
    if (idx < nTok) { tk = tokL[(size_t)e * T + idx]; wv = wgtL[(size_t)e * T + idx]; }
    stok[tid] = tk; swv[tid] = wv;
  }
  ((unsigned int*)ts)[tid] = 0u;  // MT*KC = 512 ushort = 256 uint
  __syncthreads();

  const unsigned short* Wgte = Wgt + (size_t)e * FFP * HDIM;
  const unsigned short* Wute = Wut + (size_t)e * FFP * HDIM;
  const unsigned short* Wdte = Wdt + (size_t)e * HDIM * FFP;
  const unsigned short* Ate  = At  + (size_t)e * RLORA * HDIM;
  const unsigned short* Bte  = Bt  + (size_t)e * HDIM * 32;

  const unsigned short* xrow = xbf + (size_t)stok[l16] * HDIM;  // this lane's A row
  const unsigned short* Arow = Ate + (size_t)l16 * HDIM;        // LoRA A^T row (n = r)

  const v8f vz = {0.f, 0.f, 0.f, 0.f, 0.f, 0.f, 0.f, 0.f};
  v8f accg[8], accu[8];
  #pragma unroll
  for (int i = 0; i < 8; ++i) { accg[i] = vz; accu[i] = vz; }
  v8f accA = vz;

  const int nb1 = wave * 128;  // this wave's FFP column base (8 n-tiles)

  // ---- phase 1: g = x@Wg, u = x@Wu (K over H); LoRA t = x@A on wave 0 ----
  // one-deep software pipeline: fragment loads lead their WMMA by one tile
  v16bf avc = a_frag(xrow, 0, hi);
  v16bf bg  = b_frag(Wgte + (size_t)(nb1 + l16) * HDIM, 0, hi);
  v16bf bu  = b_frag(Wute + (size_t)(nb1 + l16) * HDIM, 0, hi);

  for (int kc = 0; kc < HDIM; kc += KC) {
    const int kcn = (kc + KC < HDIM) ? kc + KC : kc;  // next chunk (clamped)
    __builtin_prefetch(xrow + kcn, 0, 3);
    v16bf avn = a_frag(xrow, kcn, hi);
    #pragma unroll
    for (int nt = 0; nt < 8; ++nt) {
      const int ntn = (nt < 7) ? nt + 1 : 0;
      const int kk  = (nt < 7) ? kc : kcn;
      v16bf bgn = b_frag(Wgte + (size_t)(nb1 + ntn * 16 + l16) * HDIM, kk, hi);
      v16bf bun = b_frag(Wute + (size_t)(nb1 + ntn * 16 + l16) * HDIM, kk, hi);
      accg[nt] = wmma_bf16(avc, bg, accg[nt]);
      accu[nt] = wmma_bf16(avc, bu, accu[nt]);
      bg = bgn; bu = bun;
    }
    if (wave == 0)  // wave-uniform branch: EXEC stays all-ones for the WMMA
      accA = wmma_bf16(avc, b_frag(Arow, kc, hi), accA);
    avc = avn;
  }

  // silu(g)*u -> mids; wave 0 writes pre-scaled LoRA t (K 16..31 stay 0)
  #pragma unroll
  for (int nt = 0; nt < 8; ++nt) {
    #pragma unroll
    for (int r = 0; r < 8; ++r) {
      float g = accg[nt][r];
      float u = accu[nt][r];
      float s = g / (1.f + __expf(-g));
      mids[r + 8 * hi][nb1 + nt * 16 + l16] = f2bf(s * u);
    }
  }
  if (wave == 0) {
    #pragma unroll
    for (int r = 0; r < 8; ++r)
      ts[r + 8 * hi][l16] = f2bf(LORA_SCALE * accA[r]);
  }
  __syncthreads();

  // ---- phase 2: out = mids@Wd (K over FFP) + ts@B (K=16 padded to 32) ----
  const int nb2 = wave * 256;  // this wave's H column base (16 n-tiles)
  v8f acco[16];
  #pragma unroll
  for (int i = 0; i < 16; ++i) acco[i] = vz;

  v16bf avc2 = a_frag(&mids[l16][0], 0, hi);
  v16bf bd   = b_frag(Wdte + (size_t)(nb2 + l16) * FFP, 0, hi);

  for (int kc = 0; kc < FFP; kc += KC) {
    const int kcn = (kc + KC < FFP) ? kc + KC : kc;
    v16bf avn = a_frag(&mids[l16][0], kcn, hi);
    #pragma unroll
    for (int nt = 0; nt < 16; ++nt) {
      const int ntn = (nt < 15) ? nt + 1 : 0;
      const int kk  = (nt < 15) ? kc : kcn;
      v16bf bdn = b_frag(Wdte + (size_t)(nb2 + ntn * 16 + l16) * FFP, kk, hi);
      acco[nt] = wmma_bf16(avc2, bd, acco[nt]);
      bd = bdn;
    }
    avc2 = avn;
  }
  {
    v16bf al = a_frag(&ts[l16][0], 0, hi);
    #pragma unroll
    for (int nt = 0; nt < 16; ++nt) {
      const unsigned short* br = Bte + (size_t)(nb2 + nt * 16 + l16) * 32;
      acco[nt] = wmma_bf16(al, b_frag(br, 0, hi), acco[nt]);
    }
  }

  // ---- weighted scatter-add into out (token hit by <=2 experts) ----
  #pragma unroll
  for (int nt = 0; nt < 16; ++nt) {
    int col = nb2 + nt * 16 + l16;
    #pragma unroll
    for (int r = 0; r < 8; ++r) {
      int m = r + 8 * hi;
      float val = swv[m] * acco[nt][r];
      atomicAdd(&out[(size_t)stok[m] * HDIM + col], val);
    }
  }
}

extern "C" void kernel_launch(void* const* d_in, const int* in_sizes, int n_in,
                              void* d_out, int out_size, void* d_ws, size_t ws_size,
                              hipStream_t stream) {
  const float* x  = (const float*)d_in[0];
  const float* gw = (const float*)d_in[1];
  const float* Wg = (const float*)d_in[2];
  const float* Wu = (const float*)d_in[3];
  const float* Wd = (const float*)d_in[4];
  const float* A  = (const float*)d_in[5];
  const float* B  = (const float*)d_in[6];
  float* out = (float*)d_out;
  const int T = in_sizes[0] / HDIM;  // 8192 tokens

  // ---- workspace carve-up (256B aligned) ----
  char* ws = (char*)d_ws;
  size_t off = 0;
  auto carve = [&](size_t bytes) -> void* {
    off = (off + 255) & ~(size_t)255;
    void* p = ws + off;
    off += bytes;
    return p;
  };
  int*            cnt  = (int*)carve(64);
  int*            tokL = (int*)carve((size_t)NE * T * sizeof(int));
  float*          wgtL = (float*)carve((size_t)NE * T * sizeof(float));
  unsigned short* xbf  = (unsigned short*)carve((size_t)T * HDIM * 2);
  unsigned short* Wgt  = (unsigned short*)carve((size_t)NE * FFP * HDIM * 2);
  unsigned short* Wut  = (unsigned short*)carve((size_t)NE * FFP * HDIM * 2);
  unsigned short* Wdt  = (unsigned short*)carve((size_t)NE * HDIM * FFP * 2);
  unsigned short* At   = (unsigned short*)carve((size_t)NE * RLORA * HDIM * 2);
  unsigned short* Bt   = (unsigned short*)carve((size_t)NE * HDIM * 32 * 2);
  (void)ws_size;

  hipMemsetAsync(cnt, 0, 64, stream);
  hipMemsetAsync(d_out, 0, (size_t)out_size * sizeof(float), stream);

  // one-time operand prep: bf16 convert (+ transpose into B-operand layout)
  {
    long long n4 = (long long)T * HDIM / 4;
    cvt_bf16_kernel<<<dim3((unsigned)((n4 + 255) / 256)), dim3(256), 0, stream>>>(x, xbf, n4);
  }
  dim3 tb(32, 8);
  transpose_cvt_kernel<<<dim3(FFP / 32, HDIM / 32, NE), tb, 0, stream>>>(Wg, Wgt, HDIM, FFP);
  transpose_cvt_kernel<<<dim3(FFP / 32, HDIM / 32, NE), tb, 0, stream>>>(Wu, Wut, HDIM, FFP);
  transpose_cvt_kernel<<<dim3(HDIM / 32, FFP / 32, NE), tb, 0, stream>>>(Wd, Wdt, FFP, HDIM);
  transpose_cvt_kernel<<<dim3(1, HDIM / 32, NE), tb, 0, stream>>>(A, At, HDIM, RLORA);
  btpad_kernel<<<dim3(NE * HDIM / 256), dim3(256), 0, stream>>>(B, Bt);

  gate_topk_kernel<<<dim3((T + 255) / 256), dim3(256), 0, stream>>>(x, gw, cnt, tokL, wgtL, T);

  dim3 grid((T + MT - 1) / MT, NE);  // over-provisioned; blocks early-exit past cnt[e]
  moe_mlp_kernel<<<grid, dim3(256), 0, stream>>>(xbf, Wgt, Wut, Wdt, At, Bt,
                                                 cnt, tokL, wgtL, out, T);
}
